// channel_attention_38697655337286
// MI455X (gfx1250) — matrix operand order repaired and checked by reading, not compiled, not run on hardware
//
#include <hip/hip_runtime.h>
#include <hip/hip_bf16.h>
#include <math.h>

#define CCH 128            // channels
#define SEQ 3000           // sequence
#define SB 128             // s-block processed per inner iteration
#define NSB 24             // ceil(SEQ/SB)
#define NSPLIT 4           // S-splits per slice in kernel1
#define SB_PER_SPLIT 6     // 4*6*128 = 3072 >= 3000
#define NSLICE 128         // 32*4
#define LD_ST 132          // padded f32 staging row stride (bank-conflict pad)
#define LD_BF 136          // padded bf16 LDS row stride (multiple of 8 -> 16B aligned runs)
#define LN_EPS 1e-5f

typedef __bf16 bf16_t;
typedef __attribute__((ext_vector_type(2)))  __bf16 v2bf;
typedef __attribute__((ext_vector_type(8)))  __bf16 v8bf;
typedef __attribute__((ext_vector_type(16))) __bf16 v16bf;
typedef __attribute__((ext_vector_type(8)))  float  v8f;

// ---- CDNA5 WMMA fragment helpers (A/B use contiguous k = 16*(lane/16) + s in both operands; C/D per ISA layout) ----

// A: 16x32 bf16, M x K. lanes 0-15: M=lane, K=k0+0..15;
// lanes 16-31: M=lane-16, K=k0+16..31. Two aligned 16B runs -> ds_load_b128 x2.
__device__ __forceinline__ v16bf load_A(const bf16_t* src, int ld, int m0, int k0, int lane) {
    int m  = m0 + (lane & 15);
    int kb = k0 + ((lane & 16) ? 16 : 0);
    const bf16_t* p = src + m * ld + kb;
    v8bf lo = *(const v8bf*)(p);
    v8bf hi = *(const v8bf*)(p + 8);
    return __builtin_shufflevector(lo, hi, 0,1,2,3,4,5,6,7,8,9,10,11,12,13,14,15);
}

// Swizzled (fragment-major) B storage: element (k,n), k,n in [0,128):
// idx = [(((n>>4)*4 + (k>>5))*2 + ((k>>4)&1))*16 + (n&15)]*16 + (k&15)
__device__ __forceinline__ int bsw_idx(int k, int n) {
    return (((((n >> 4) * 4 + (k >> 5)) * 2 + ((k >> 4) & 1)) * 16 + (n & 15)) * 16) + (k & 15);
}

// B: 32x16 fragment (k-tile kk, n-tile nt): each lane reads 16 contiguous halves (32B aligned).
__device__ __forceinline__ v16bf load_B_sw(const bf16_t* Bsw, int nt, int kk, int lane) {
    int base = (((nt * 4 + kk) * 2 + (lane >> 4)) * 16 + (lane & 15)) * 16;
    return *(const v16bf*)(Bsw + base);
}

// C/D: 16x16 f32. reg r: lanes 0-15 -> M=m0+r, lanes 16-31 -> M=m0+8+r; N=n0+(lane&15).
__device__ __forceinline__ void store_C_bias(float* dst, int ld, int m0, int n0, int lane,
                                             v8f c, const float* __restrict__ bias) {
    int col = n0 + (lane & 15);
    int rb  = m0 + ((lane >> 4) << 3);
    float bv = bias[col];
#pragma unroll
    for (int r = 0; r < 8; ++r) dst[(rb + r) * ld + col] = c[r] + bv;
}

// ---- block-wide LayerNorm statistics over 128 rows x 128 cols in f32 staging ----
// red layout: psum[2*128] | psq[2*128] | mean[128] | rstd[128]  (768 floats)
__device__ __forceinline__ void block_ln_stats(const float* stag, float* red, int tid) {
    int row = tid & 127, half = tid >> 7;
    const float* rp = stag + row * LD_ST + half * 64;
    float s = 0.f, q = 0.f;
#pragma unroll 8
    for (int i = 0; i < 64; ++i) { float v = rp[i]; s += v; q += v * v; }
    red[half * 128 + row]       = s;
    red[256 + half * 128 + row] = q;
    __syncthreads();
    if (tid < 128) {
        float ss = red[tid] + red[128 + tid];
        float qq = red[256 + tid] + red[384 + tid];
        float m  = ss * (1.0f / 128.0f);
        float var = qq * (1.0f / 128.0f) - m * m;
        red[512 + tid] = m;
        red[640 + tid] = rsqrtf(var + LN_EPS);
    }
    __syncthreads();
}

// ---- kernel 0: zero atten accumulator ----
__global__ void k_zero(float* p, int n) {
    int i = blockIdx.x * blockDim.x + threadIdx.x;
    if (i < n) p[i] = 0.f;
}

// ---- kernel 1: per (slice, S-split): Q/K linear+LN in LDS, accumulate atten ----
__global__ void __launch_bounds__(256)
k_qk_atten(const float* __restrict__ x,
           const float* __restrict__ Wq, const float* __restrict__ bq,
           const float* __restrict__ gq, const float* __restrict__ betaq,
           const float* __restrict__ Wk, const float* __restrict__ bk,
           const float* __restrict__ gk, const float* __restrict__ betak,
           float* __restrict__ atten)
{
    extern __shared__ char smem[];
    float*  stag = (float*)smem;                       // 128*LD_ST f32 (weights staged here first)
    bf16_t* temp = (bf16_t*)(smem + 128 * LD_ST * 4);  // x tile [s][c]; later aliased as swizzled K
    bf16_t* qT   = temp + 128 * LD_BF;                 // q LN'd, [c'][s]
    float*  red  = (float*)(qT + 128 * LD_BF);         // 768 floats
    bf16_t* ksw  = temp;                               // swizzled K operand (alias, 16384 elems)

    const int tid  = threadIdx.x;
    const int lane = tid & 31, wave = tid >> 5;
    const int ws = wave >> 1;      // 0..3 -> row-tiles {2ws, 2ws+1}
    const int wn = wave & 1;       // 0..1 -> col-tiles 4wn..4wn+3
    const int slice = blockIdx.y;
    const int split = blockIdx.x;
    const float* xs = x + (size_t)slice * CCH * SEQ;

    // stage weights into swizzled LDS (aliased onto stag), then hoist to registers
    {
        bf16_t* wqsw = (bf16_t*)stag;
        bf16_t* wksw = wqsw + 16384;
        for (int e = tid; e < 128 * 128; e += 256) {
            int cp = e >> 7, c = e & 127;          // Wq[c'][c]: B element (k=c, n=c')
            int idx = bsw_idx(c, cp);
            wqsw[idx] = (bf16_t)Wq[e];
            wksw[idx] = (bf16_t)Wk[e];
        }
        __syncthreads();
    }
    v16bf wqf[4][4], wkf[4][4];    // [j: local n-tile][kk]
#pragma unroll
    for (int j = 0; j < 4; ++j)
#pragma unroll
        for (int kk = 0; kk < 4; ++kk) {
            wqf[j][kk] = load_B_sw((bf16_t*)stag,         4 * wn + j, kk, lane);
            wkf[j][kk] = load_B_sw((bf16_t*)stag + 16384, 4 * wn + j, kk, lane);
        }
    __syncthreads();   // weights consumed; stag free for f32 staging

    v8f acc[2][4];
#pragma unroll
    for (int i = 0; i < 2; ++i)
#pragma unroll
        for (int j = 0; j < 4; ++j) acc[i][j] = (v8f){0.f,0.f,0.f,0.f,0.f,0.f,0.f,0.f};

    for (int ib = 0; ib < SB_PER_SPLIT; ++ib) {
        const int s0 = (split * SB_PER_SPLIT + ib) * SB;

        // x tile -> temp[s][c] bf16 pairs (zero padded beyond SEQ); coalesced along s
        for (int e = tid; e < 64 * SB; e += 256) {
            int c2 = e >> 7, j = e & 127, c = c2 * 2;
            bool vld = (s0 + j) < SEQ;
            float va = vld ? xs[(size_t)c * SEQ + s0 + j] : 0.f;
            float vb = vld ? xs[(size_t)(c + 1) * SEQ + s0 + j] : 0.f;
            *(v2bf*)(temp + j * LD_BF + c) = (v2bf){(bf16_t)va, (bf16_t)vb};
        }
        if (ib + 1 < SB_PER_SPLIT) {
            int sn = s0 + SB;
            if (tid < CCH && sn < SEQ)
                __builtin_prefetch(&xs[(size_t)tid * SEQ + sn], 0, 1);
        }
        __syncthreads();

        // Q = temp @ Wq^T + bq -> stag
        {
            v16bf a[2][4];
#pragma unroll
            for (int i = 0; i < 2; ++i)
#pragma unroll
                for (int kk = 0; kk < 4; ++kk)
                    a[i][kk] = load_A(temp, LD_BF, 16 * (2 * ws + i), 32 * kk, lane);
#pragma unroll
            for (int i = 0; i < 2; ++i)
#pragma unroll
                for (int j = 0; j < 4; ++j) {
                    v8f cf = {};
#pragma unroll
                    for (int kk = 0; kk < 4; ++kk)
                        cf = __builtin_amdgcn_wmma_f32_16x16x32_bf16(false, a[i][kk], false,
                                 wqf[j][kk], (short)0, cf, false, false);
                    store_C_bias(stag, LD_ST, 16 * (2 * ws + i), 16 * (4 * wn + j), lane, cf, bq);
                }
        }
        __syncthreads();
        block_ln_stats(stag, red, tid);
        {   // normalize -> qT[c'][s], packed pairs along s
            int p = tid & 63, qq = tid >> 6;
            int r0 = 2 * p, r1 = r0 + 1;
            bool v0 = (s0 + r0) < SEQ, v1 = (s0 + r1) < SEQ;
            float m0 = red[512 + r0], rr0 = red[640 + r0];
            float m1 = red[512 + r1], rr1 = red[640 + r1];
            const float* rp0 = stag + r0 * LD_ST;
            const float* rp1 = stag + r1 * LD_ST;
#pragma unroll 4
            for (int i = 0; i < 32; ++i) {
                int c = 32 * qq + i;
                float a0 = v0 ? (rp0[c] - m0) * rr0 * gq[c] + betaq[c] : 0.f;
                float a1 = v1 ? (rp1[c] - m1) * rr1 * gq[c] + betaq[c] : 0.f;
                *(v2bf*)(qT + c * LD_BF + r0) = (v2bf){(bf16_t)a0, (bf16_t)a1};
            }
        }
        __syncthreads();

        // K = temp @ Wk^T + bk -> stag
        {
            v16bf a[2][4];
#pragma unroll
            for (int i = 0; i < 2; ++i)
#pragma unroll
                for (int kk = 0; kk < 4; ++kk)
                    a[i][kk] = load_A(temp, LD_BF, 16 * (2 * ws + i), 32 * kk, lane);
#pragma unroll
            for (int i = 0; i < 2; ++i)
#pragma unroll
                for (int j = 0; j < 4; ++j) {
                    v8f cf = {};
#pragma unroll
                    for (int kk = 0; kk < 4; ++kk)
                        cf = __builtin_amdgcn_wmma_f32_16x16x32_bf16(false, a[i][kk], false,
                                 wkf[j][kk], (short)0, cf, false, false);
                    store_C_bias(stag, LD_ST, 16 * (2 * ws + i), 16 * (4 * wn + j), lane, cf, bk);
                }
        }
        __syncthreads();
        block_ln_stats(stag, red, tid);
        {   // normalize -> swizzled K operand (aliased onto temp; temp now dead)
            int p = tid & 63, qq = tid >> 6;
            int r0 = 2 * p, r1 = r0 + 1;
            bool v0 = (s0 + r0) < SEQ, v1 = (s0 + r1) < SEQ;
            float m0 = red[512 + r0], rr0 = red[640 + r0];
            float m1 = red[512 + r1], rr1 = red[640 + r1];
            const float* rp0 = stag + r0 * LD_ST;
            const float* rp1 = stag + r1 * LD_ST;
#pragma unroll 4
            for (int i = 0; i < 32; ++i) {
                int c = 32 * qq + i;   // m index; k index = s (r0 even -> pair contiguous)
                float a0 = v0 ? (rp0[c] - m0) * rr0 * gk[c] + betak[c] : 0.f;
                float a1 = v1 ? (rp1[c] - m1) * rr1 * gk[c] + betak[c] : 0.f;
                *(v2bf*)(ksw + bsw_idx(r0, c)) = (v2bf){(bf16_t)a0, (bf16_t)a1};
            }
        }
        __syncthreads();

        // atten += q[c,s] * k[m,s] : A = qT[c][s], B = swizzled K
        {
            v16bf a[2][4];
#pragma unroll
            for (int i = 0; i < 2; ++i)
#pragma unroll
                for (int kk = 0; kk < 4; ++kk)
                    a[i][kk] = load_A(qT, LD_BF, 16 * (2 * ws + i), 32 * kk, lane);
#pragma unroll
            for (int j = 0; j < 4; ++j) {
                v16bf bfr[4];
#pragma unroll
                for (int kk = 0; kk < 4; ++kk) bfr[kk] = load_B_sw(ksw, 4 * wn + j, kk, lane);
#pragma unroll
                for (int i = 0; i < 2; ++i)
#pragma unroll
                    for (int kk = 0; kk < 4; ++kk)
                        acc[i][j] = __builtin_amdgcn_wmma_f32_16x16x32_bf16(false, a[i][kk], false,
                                        bfr[kk], (short)0, acc[i][j], false, false);
            }
        }
        __syncthreads();
    }

    // flush partial atten with f32 global atomics
    float* ap = atten + (size_t)slice * CCH * CCH;
    int col = lane & 15, rb = (lane >> 4) << 3;
#pragma unroll
    for (int i = 0; i < 2; ++i)
#pragma unroll
        for (int j = 0; j < 4; ++j) {
            int row0 = 16 * (2 * ws + i) + rb;
            int c0   = 16 * (4 * wn + j) + col;
#pragma unroll
            for (int r = 0; r < 8; ++r)
                atomicAdd(&ap[(row0 + r) * CCH + c0], acc[i][j][r]);
        }
}

// ---- kernel 2: softmax row-sum (rows sum to ~1, kept faithful) ----
__global__ void k_softmax_rowsum(const float* __restrict__ atten, float* __restrict__ rowsum) {
    int slice = blockIdx.x, c = threadIdx.x;
    const float* a = atten + (size_t)slice * CCH * CCH + (size_t)c * CCH;
    const float inv_scale = 0.1f;   // 1/sqrt(extract=100)
    float mx = -1e30f;
    for (int i = 0; i < CCH; ++i) mx = fmaxf(mx, a[i] * inv_scale);
    float s = 0.f;
    for (int i = 0; i < CCH; ++i) s += __expf(a[i] * inv_scale - mx);
    float inv = 1.f / s, t = 0.f;
    for (int i = 0; i < CCH; ++i) t += __expf(a[i] * inv_scale - mx) * inv;
    rowsum[slice * CCH + c] = t;
}

// ---- kernel 3: out = LN((x*rowsum)^T @ Wp^T + bp), stored transposed ----
__global__ void __launch_bounds__(256)
k_proj(const float* __restrict__ x,
       const float* __restrict__ Wp, const float* __restrict__ bp,
       const float* __restrict__ gp, const float* __restrict__ betap,
       const float* __restrict__ rowsum, float* __restrict__ out)
{
    extern __shared__ char smem[];
    float*  stag = (float*)smem;                       // 128*LD_ST f32 (weight staged here first)
    bf16_t* temp = (bf16_t*)(smem + 128 * LD_ST * 4);  // [s][c]
    float*  rs   = (float*)(temp + 128 * LD_BF);       // 128
    float*  red  = rs + 128;                           // 768

    const int tid  = threadIdx.x;
    const int lane = tid & 31, wave = tid >> 5;
    const int ws = wave >> 1, wn = wave & 1;
    const int slice = blockIdx.y;
    const int s0 = blockIdx.x * SB;
    const float* xs = x + (size_t)slice * CCH * SEQ;
    float* os = out + (size_t)slice * CCH * SEQ;

    {
        bf16_t* wpsw = (bf16_t*)stag;
        for (int e = tid; e < 128 * 128; e += 256) {
            int cp = e >> 7, c = e & 127;
            wpsw[bsw_idx(c, cp)] = (bf16_t)Wp[e];
        }
        if (tid < 128) rs[tid] = rowsum[slice * CCH + tid];
        __syncthreads();
    }
    v16bf wpf[4][4];
#pragma unroll
    for (int j = 0; j < 4; ++j)
#pragma unroll
        for (int kk = 0; kk < 4; ++kk)
            wpf[j][kk] = load_B_sw((bf16_t*)stag, 4 * wn + j, kk, lane);
    __syncthreads();

    for (int e = tid; e < 64 * SB; e += 256) {
        int c2 = e >> 7, j = e & 127, c = c2 * 2;
        bool vld = (s0 + j) < SEQ;
        float va = vld ? xs[(size_t)c * SEQ + s0 + j] * rs[c] : 0.f;
        float vb = vld ? xs[(size_t)(c + 1) * SEQ + s0 + j] * rs[c + 1] : 0.f;
        *(v2bf*)(temp + j * LD_BF + c) = (v2bf){(bf16_t)va, (bf16_t)vb};
    }
    __syncthreads();

    {
        v16bf a[2][4];
#pragma unroll
        for (int i = 0; i < 2; ++i)
#pragma unroll
            for (int kk = 0; kk < 4; ++kk)
                a[i][kk] = load_A(temp, LD_BF, 16 * (2 * ws + i), 32 * kk, lane);
#pragma unroll
        for (int i = 0; i < 2; ++i)
#pragma unroll
            for (int j = 0; j < 4; ++j) {
                v8f cf = {};
#pragma unroll
                for (int kk = 0; kk < 4; ++kk)
                    cf = __builtin_amdgcn_wmma_f32_16x16x32_bf16(false, a[i][kk], false,
                             wpf[j][kk], (short)0, cf, false, false);
                store_C_bias(stag, LD_ST, 16 * (2 * ws + i), 16 * (4 * wn + j), lane, cf, bp);
            }
    }
    __syncthreads();
    block_ln_stats(stag, red, tid);
    {   // normalize and store transposed: out[c][s0+row]; coalesced over rows
        int row = tid & 127, half = tid >> 7;
        bool valid = (s0 + row) < SEQ;
        float m = red[512 + row], r = red[640 + row];
        const float* rp = stag + row * LD_ST + half * 64;
#pragma unroll 4
        for (int i = 0; i < 64; ++i) {
            int c = half * 64 + i;
            float v = (rp[i] - m) * r * gp[c] + betap[c];
            if (valid) os[(size_t)c * SEQ + s0 + row] = v;
        }
    }
}

extern "C" void kernel_launch(void* const* d_in, const int* in_sizes, int n_in,
                              void* d_out, int out_size, void* d_ws, size_t ws_size,
                              hipStream_t stream) {
    const float* x     = (const float*)d_in[0];
    const float* Wq    = (const float*)d_in[1];
    const float* bq    = (const float*)d_in[2];
    const float* gq    = (const float*)d_in[3];
    const float* betaq = (const float*)d_in[4];
    const float* Wk    = (const float*)d_in[5];
    const float* bk    = (const float*)d_in[6];
    const float* gk    = (const float*)d_in[7];
    const float* betak = (const float*)d_in[8];
    const float* Wp    = (const float*)d_in[9];
    const float* bp    = (const float*)d_in[10];
    const float* gp    = (const float*)d_in[11];
    const float* betap = (const float*)d_in[12];
    float* out = (float*)d_out;

    float* atten  = (float*)d_ws;                 // 128*128*128 f32 = 8 MB
    float* rowsum = atten + NSLICE * CCH * CCH;   // 64 KB

    const size_t smem1 = 128 * LD_ST * 4 + 2 * 128 * LD_BF * 2 + 768 * 4;            // ~137 KB
    const size_t smem3 = 128 * LD_ST * 4 + 128 * LD_BF * 2 + 128 * 4 + 768 * 4;      // ~104 KB
    hipFuncSetAttribute(reinterpret_cast<const void*>(k_qk_atten),
                        hipFuncAttributeMaxDynamicSharedMemorySize, (int)smem1);
    hipFuncSetAttribute(reinterpret_cast<const void*>(k_proj),
                        hipFuncAttributeMaxDynamicSharedMemorySize, (int)smem3);

    const int nz = NSLICE * CCH * CCH;
    k_zero<<<(nz + 255) / 256, 256, 0, stream>>>(atten, nz);
    k_qk_atten<<<dim3(NSPLIT, NSLICE), 256, smem1, stream>>>(
        x, Wq, bq, gq, betaq, Wk, bk, gk, betak, atten);
    k_softmax_rowsum<<<NSLICE, CCH, 0, stream>>>(atten, rowsum);
    k_proj<<<dim3(NSB, NSLICE), 256, smem3, stream>>>(
        x, Wp, bp, gp, betap, rowsum, out);
}
